// arch2_70978629534162
// MI455X (gfx1250) — matrix operand order repaired and checked by reading, not compiled, not run on hardware
//
#include <hip/hip_runtime.h>

#define NPTS 2048
#define DMODEL 256
#define HEADS 8
#define KSZ 32
#define FFD 1024
#define LAYERS 16
#define LN_EPS 1e-5f

typedef __attribute__((ext_vector_type(16))) __bf16 v16bf;
typedef __attribute__((ext_vector_type(8)))  float  v8f;

union BF16x16 { v16bf v; unsigned short u[16]; uint4 q[2]; };

// ---------- bf16 helpers (round-to-nearest-even) ----------
__device__ __forceinline__ unsigned short f2bf(float f) {
  union { float f; unsigned u; } x; x.f = f;
  unsigned r = x.u + 0x7FFFu + ((x.u >> 16) & 1u);
  return (unsigned short)(r >> 16);
}
__device__ __forceinline__ float bf2f(unsigned short h) {
  union { float f; unsigned u; } x; x.u = ((unsigned)h) << 16;
  return x.f;
}
__device__ __forceinline__ float gelu_tanh(float x) {
  float x3 = x * x * x;
  return 0.5f * x * (1.f + tanhf(0.7978845608028654f * (x + 0.044715f * x3)));
}

// ---------- WMMA ----------
__device__ __forceinline__ v8f wmma_bf16(v16bf a, v16bf b, v8f c) {
  return __builtin_amdgcn_wmma_f32_16x16x32_bf16(false, a, false, b, (short)0, c, false, false);
}

// A fragment (16x32 bf16, row-major src): lane M=l%16; K=e+16*half
__device__ __forceinline__ v16bf load_a_frag(const unsigned short* A, int lda, int m0, int k0) {
  int lane = threadIdx.x & 31, half = lane >> 4;
  const unsigned short* p = A + (size_t)(m0 + (lane & 15)) * lda + k0;
  BF16x16 f;
  f.q[0] = *(const uint4*)(p + 16 * half);
  f.q[1] = *(const uint4*)(p + 16 * half + 8);
  return f.v;
}
// B fragment where B[kk,n] = Src[base_row+n, k0+kk] (row-major bf16 src): lane col n=l%16, K=e+16*half
__device__ __forceinline__ v16bf load_bT_frag(const unsigned short* S, int lds_, int base_row, int k0) {
  int lane = threadIdx.x & 31;
  const unsigned short* p = S + (size_t)(base_row + (lane & 15)) * lds_ + k0 + 16 * (lane >> 4);
  BF16x16 f;
  f.q[0] = *(const uint4*)(p);
  f.q[1] = *(const uint4*)(p + 8);
  return f.v;
}
// B fragment from an LDS tile [32 rows x 32 kk] (row n computed per-lane by caller)
__device__ __forceinline__ v16bf lds_frag(const unsigned short* tile, int n) {
  int half = (threadIdx.x >> 4) & 1;
  const unsigned short* p = tile + n * 32 + 16 * half;
  BF16x16 f;
  f.q[0] = *(const uint4*)(p);
  f.q[1] = *(const uint4*)(p + 8);
  return f.v;
}

// ---------- permlane16 xor-reductions (VALU cross-lane; row of 16) ----------
constexpr unsigned selword(int base, int m) {
  unsigned s = 0;
  for (int i = 0; i < 8; ++i) s |= (unsigned)(((base + i) ^ m) & 0xF) << (4 * i);
  return s;
}
template <int M>
__device__ __forceinline__ float xor16(float x) {
  int i = __float_as_int(x);
  i = __builtin_amdgcn_permlane16(i, i, selword(0, M), selword(8, M), false, false);
  return __int_as_float(i);
}
__device__ __forceinline__ float rowmax16(float m) {
  m = fmaxf(m, xor16<1>(m)); m = fmaxf(m, xor16<2>(m));
  m = fmaxf(m, xor16<4>(m)); m = fmaxf(m, xor16<8>(m));
  return m;
}
__device__ __forceinline__ float rowsum16(float s) {
  s += xor16<1>(s); s += xor16<2>(s); s += xor16<4>(s); s += xor16<8>(s);
  return s;
}

// ---------- async global -> LDS (gfx1250, tracked with ASYNCcnt) ----------
__device__ __forceinline__ unsigned to_lds_off(const void* p) {
  return (unsigned)(unsigned long long)(__attribute__((address_space(3))) const char*)p;
}
__device__ __forceinline__ void async_b128(unsigned ldsoff, const void* g) {
  asm volatile("global_load_async_to_lds_b128 %0, %1, off"
               :: "v"(ldsoff), "v"((unsigned long long)(size_t)g) : "memory");
}
__device__ __forceinline__ void wait_async_le8() {
  asm volatile("s_wait_asynccnt 0x8" ::: "memory");
}

// ---------- small kernels ----------
__global__ void embed_kernel(const float* __restrict__ x, const float* __restrict__ W1,
                             const float* __restrict__ b1, const float* __restrict__ W2,
                             const float* __restrict__ b2, float* __restrict__ z) {
  int idx = blockIdx.x * 256 + threadIdx.x;
  int n = idx >> 8, d = idx & 255;
  float acc;
  if (d < 128) {
    acc = b1[d];
#pragma unroll
    for (int f = 0; f < 12; ++f) acc += x[n * 13 + f] * W1[f * 128 + d];
  } else {
    acc = b2[d - 128] + x[n * 13 + 12] * W2[d - 128];
  }
  z[(size_t)n * DMODEL + d] = acc;
}

__device__ __forceinline__ float block_sum256(float v, float* red) {
#pragma unroll
  for (int off = 16; off > 0; off >>= 1) v += __shfl_xor(v, off, 32);
  int wid = threadIdx.x >> 5;
  if ((threadIdx.x & 31) == 0) red[wid] = v;
  __syncthreads();
  float tot = 0.f;
#pragma unroll
  for (int i = 0; i < 8; ++i) tot += red[i];
  __syncthreads();
  return tot;
}

// LayerNorm row kernel -> bf16 output
__global__ void ln_kernel(const float* __restrict__ z, const float* __restrict__ s,
                          const float* __restrict__ o, unsigned short* __restrict__ out) {
  __shared__ float red[8];
  int r = blockIdx.x, t = threadIdx.x;
  float x = z[(size_t)r * DMODEL + t];
  float sum = block_sum256(x, red);
  float sq  = block_sum256(x * x, red);
  float mean = sum * (1.f / DMODEL);
  float var  = sq * (1.f / DMODEL) - mean * mean;
  float y = (x - mean) * rsqrtf(var + LN_EPS) * s[t] + o[t];
  out[(size_t)r * DMODEL + t] = f2bf(y);
}

// transpose + convert weights: W[Kd x Nd] f32 -> WT[Nd x Kd] bf16
__global__ void wconv_kernel(const float* __restrict__ W, unsigned short* __restrict__ WT,
                             int Kd, int Nd) {
  int idx = blockIdx.x * 256 + threadIdx.x;
  int n = idx / Kd, k = idx - n * Kd;
  WT[(size_t)n * Kd + k] = f2bf(W[(size_t)k * Nd + n]);
}

// per-(point,head) squared norm of k
__global__ void k2_kernel(const unsigned short* __restrict__ kb, float* __restrict__ k2) {
  int idx = blockIdx.x * 256 + threadIdx.x;  // N*H = 16384
  int n = idx >> 3, h = idx & 7;
  const unsigned short* p = kb + (size_t)n * DMODEL + h * KSZ;
  float s = 0.f;
#pragma unroll
  for (int j = 0; j < KSZ; ++j) { float v = bf2f(p[j]); s += v * v; }
  k2[idx] = s;
}

// final: LN -> LN -> dot W_sc
__global__ void final_kernel(const float* __restrict__ z, const float* __restrict__ s1,
                             const float* __restrict__ o1, const float* __restrict__ s2,
                             const float* __restrict__ o2, const float* __restrict__ Wsc,
                             const float* __restrict__ bsc, float* __restrict__ out) {
  __shared__ float red[8];
  int r = blockIdx.x, t = threadIdx.x;
  float x = z[(size_t)r * DMODEL + t];
  float sum = block_sum256(x, red), sq = block_sum256(x * x, red);
  float mean = sum * (1.f / DMODEL), var = sq * (1.f / DMODEL) - mean * mean;
  float y = (x - mean) * rsqrtf(var + LN_EPS) * s1[t] + o1[t];
  float sum2 = block_sum256(y, red), sq2 = block_sum256(y * y, red);
  float m2 = sum2 * (1.f / DMODEL), v2 = sq2 * (1.f / DMODEL) - m2 * m2;
  float yn = (y - m2) * rsqrtf(v2 + LN_EPS) * s2[t] + o2[t];
  float tot = block_sum256(yn * Wsc[t], red);
  if (t == 0) out[r] = tot + bsc[0];
}

// ---------- WMMA GEMM: C[M,Nd] = op(A_bf16[M,Kd] @ BT_bf16[Nd,Kd]^T + bias) ----------
// MODE 0: bf16 out; MODE 1: gelu -> bf16 out; MODE 2: f32 residual accumulate; MODE 3: bf16 transposed out
template <int MODE>
__global__ __launch_bounds__(256) void gemm_kernel(const unsigned short* __restrict__ A,
                                                   const unsigned short* __restrict__ BT,
                                                   const float* __restrict__ bias,
                                                   void* __restrict__ Cout,
                                                   int M, int Nd, int Kd) {
  int gw = blockIdx.x * 8 + (threadIdx.x >> 5);
  int tiles_n = Nd >> 6;
  int m0 = (gw / tiles_n) * 16;
  int n0 = (gw % tiles_n) * 64;
  if (m0 >= M) return;

  v8f acc[4];
#pragma unroll
  for (int t = 0; t < 4; ++t)
#pragma unroll
    for (int i = 0; i < 8; ++i) acc[t][i] = 0.f;

  for (int k0 = 0; k0 < Kd; k0 += 32) {
    v16bf a = load_a_frag(A, Kd, m0, k0);
    if (k0 + 32 < Kd)  // prefetch next weight tile into cache (global_prefetch_b8)
      __builtin_prefetch(BT + (size_t)(n0 + (threadIdx.x & 31)) * Kd + k0 + 32, 0, 3);
#pragma unroll
    for (int t = 0; t < 4; ++t) {
      v16bf b = load_bT_frag(BT, Kd, n0 + 16 * t, k0);
      acc[t] = wmma_bf16(a, b, acc[t]);
    }
  }

  int lane = threadIdx.x & 31, half = lane >> 4, lnn = lane & 15;
#pragma unroll
  for (int t = 0; t < 4; ++t) {
    int col = n0 + 16 * t + lnn;
    float bv = bias[col];
#pragma unroll
    for (int v = 0; v < 8; ++v) {
      int row = m0 + v + 8 * half;
      float val = acc[t][v] + bv;
      if (MODE == 0)
        ((unsigned short*)Cout)[(size_t)row * Nd + col] = f2bf(val);
      else if (MODE == 1)
        ((unsigned short*)Cout)[(size_t)row * Nd + col] = f2bf(gelu_tanh(val));
      else if (MODE == 2)
        ((float*)Cout)[(size_t)row * Nd + col] += val;
      else
        ((unsigned short*)Cout)[(size_t)col * M + row] = f2bf(val);  // transposed (vT)
    }
  }
}

// ---------- flash attention: one wave per (head, 16-query tile) ----------
// K/V tiles stream into LDS via global_load_async_to_lds_b128, double-buffered.
__global__ __launch_bounds__(256) void attn_kernel(const unsigned short* __restrict__ qb,
                                                   const unsigned short* __restrict__ kb,
                                                   const unsigned short* __restrict__ vT,
                                                   const float* __restrict__ k2,
                                                   unsigned short* __restrict__ ob) {
  __shared__ __align__(16) unsigned short kl[8][2][32 * 32];  // 32 keys x 32 dims per buf
  __shared__ __align__(16) unsigned short vl[8][2][32 * 32];  // 32 dims x 32 keys per buf
  __shared__ __align__(16) unsigned short pls[8][16 * 32];    // per-wave P tile (bf16)
  int wid = threadIdx.x >> 5;
  int lane = threadIdx.x & 31, half = lane >> 4, lnn = lane & 15;
  int gw = blockIdx.x * 8 + wid;
  int h = gw >> 7;                 // 8 heads
  int q0 = (gw & 127) << 4;        // 128 query tiles of 16
  const float scale = 0.17677669529663687f;  // 1/sqrt(32)

  v16bf aq = load_a_frag(qb, DMODEL, q0, h * KSZ);
  v8f o1, o2;
  float mrow[8], lrow[8];
#pragma unroll
  for (int v = 0; v < 8; ++v) { o1[v] = 0.f; o2[v] = 0.f; mrow[v] = -1e30f; lrow[v] = 0.f; }

  // stage chunk (32 keys) into per-wave LDS tiles: one b128 x4 per lane per tile
  auto stage = [&](int buf, int c) {
    const unsigned short* kg = kb + (size_t)(c + lane) * DMODEL + h * KSZ;
    const unsigned short* vg = vT + (size_t)(h * KSZ + lane) * NPTS + c;
    unsigned kd = to_lds_off(&kl[wid][buf][lane * 32]);
    unsigned vd = to_lds_off(&vl[wid][buf][lane * 32]);
#pragma unroll
    for (int i = 0; i < 4; ++i) {
      async_b128(kd + 16 * i, kg + 8 * i);
      async_b128(vd + 16 * i, vg + 8 * i);
    }
  };

  stage(0, 0);
  int buf = 0;
  for (int c = 0; c < NPTS; c += 32) {
    stage(buf ^ 1, (c + 32) & (NPTS - 1));  // prefetch next (wraps harmlessly on last iter)
    wait_async_le8();                        // current buffer's 8 async loads complete

    v16bf b1 = lds_frag(&kl[wid][buf][0], lnn);
    v16bf b2 = lds_frag(&kl[wid][buf][0], 16 + lnn);
    v8f zero;
#pragma unroll
    for (int i = 0; i < 8; ++i) zero[i] = 0.f;
    v8f s1 = wmma_bf16(aq, b1, zero);
    v8f s2 = wmma_bf16(aq, b2, zero);
    float kk1 = k2[(c + lnn) * HEADS + h];
    float kk2 = k2[(c + 16 + lnn) * HEADS + h];

#pragma unroll
    for (int v = 0; v < 8; ++v) {
      // logits' = 2s*qk - s*k2  (the -s*q2 term cancels in softmax)
      float l1 = 2.f * scale * s1[v] - scale * kk1;
      float l2 = 2.f * scale * s2[v] - scale * kk2;
      float m = rowmax16(fmaxf(l1, l2));
      float newm = fmaxf(mrow[v], m);
      float fac = __expf(mrow[v] - newm);
      float p1 = __expf(l1 - newm);
      float p2 = __expf(l2 - newm);
      float rs = rowsum16(p1 + p2);
      lrow[v] = lrow[v] * fac + rs;
      mrow[v] = newm;
      o1[v] *= fac; o2[v] *= fac;
      int prow = v + 8 * half;                    // C layout -> LDS row-major P
      pls[wid][prow * 32 + lnn]      = f2bf(p1);
      pls[wid][prow * 32 + 16 + lnn] = f2bf(p2);
    }
    __syncthreads();  // uniform trip count across all 8 waves

    // reload P as an A fragment (16x32)
    BF16x16 fa;
    const unsigned short* pp = &pls[wid][lnn * 32];
    fa.q[0] = *(const uint4*)(pp + 16 * half);
    fa.q[1] = *(const uint4*)(pp + 16 * half + 8);

    v16bf bv1 = lds_frag(&vl[wid][buf][0], lnn);
    v16bf bv2 = lds_frag(&vl[wid][buf][0], 16 + lnn);
    o1 = wmma_bf16(fa.v, bv1, o1);
    o2 = wmma_bf16(fa.v, bv2, o2);
    __syncthreads();
    buf ^= 1;
  }

#pragma unroll
  for (int v = 0; v < 8; ++v) {
    float inv = 1.f / lrow[v];
    int row = q0 + v + 8 * half;
    ob[(size_t)row * DMODEL + h * KSZ + lnn]      = f2bf(o1[v] * inv);
    ob[(size_t)row * DMODEL + h * KSZ + 16 + lnn] = f2bf(o2[v] * inv);
  }
}

// ---------- host ----------
extern "C" void kernel_launch(void* const* d_in, const int* in_sizes, int n_in,
                              void* d_out, int out_size, void* d_ws, size_t ws_size,
                              hipStream_t stream) {
  (void)in_sizes; (void)n_in; (void)out_size; (void)ws_size;
  const float* x      = (const float*)d_in[0];
  const float* W_in1  = (const float*)d_in[1];
  const float* b_in1  = (const float*)d_in[2];
  const float* W_in2  = (const float*)d_in[3];
  const float* b_in2  = (const float*)d_in[4];
  const float* ln_q_s = (const float*)d_in[5];
  const float* ln_q_o = (const float*)d_in[6];
  const float* ln_k_s = (const float*)d_in[7];
  const float* ln_k_o = (const float*)d_in[8];
  const float* ln_v_s = (const float*)d_in[9];
  const float* ln_v_o = (const float*)d_in[10];
  const float* Wq  = (const float*)d_in[11];
  const float* bq  = (const float*)d_in[12];
  const float* Wk  = (const float*)d_in[13];
  const float* bk  = (const float*)d_in[14];
  const float* Wv  = (const float*)d_in[15];
  const float* bv  = (const float*)d_in[16];
  const float* Wo  = (const float*)d_in[17];
  const float* bo  = (const float*)d_in[18];
  const float* ln_f_s = (const float*)d_in[19];
  const float* ln_f_o = (const float*)d_in[20];
  const float* Wf1 = (const float*)d_in[21];
  const float* bf1 = (const float*)d_in[22];
  const float* Wf2 = (const float*)d_in[23];
  const float* bf2 = (const float*)d_in[24];
  const float* ln_out_s = (const float*)d_in[25];
  const float* ln_out_o = (const float*)d_in[26];
  const float* ln_sc_s  = (const float*)d_in[27];
  const float* ln_sc_o  = (const float*)d_in[28];
  const float* W_sc = (const float*)d_in[29];
  const float* b_sc = (const float*)d_in[30];

  const size_t MB = (size_t)1 << 20;
  char* ws = (char*)d_ws;
  float*          z    = (float*)(ws + 0 * MB);            // 2 MB
  unsigned short* a_bf = (unsigned short*)(ws + 2 * MB);   // 1 MB  (LN output)
  unsigned short* q_bf = (unsigned short*)(ws + 3 * MB);   // 1 MB
  unsigned short* k_bf = (unsigned short*)(ws + 4 * MB);   // 1 MB
  unsigned short* v_bf = (unsigned short*)(ws + 5 * MB);   // 1 MB  (vT [DMODEL][NPTS])
  float*          k2b  = (float*)(ws + 6 * MB);            // 64 KB
  unsigned short* o_bf = (unsigned short*)(ws + 7 * MB);   // 1 MB
  unsigned short* h_bf = (unsigned short*)(ws + 8 * MB);   // 4 MB
  unsigned short* wT   = (unsigned short*)(ws + 12 * MB);  // 512 KB (per-use transposed weights)

  const int g256  = (NPTS / 16) * (DMODEL / 64) / 8;  // 64 blocks
  const int g1024 = (NPTS / 16) * (FFD / 64) / 8;     // 256 blocks
  const int cvt256  = DMODEL * DMODEL / 256;          // 256 blocks
  const int cvt1024 = DMODEL * FFD / 256;             // 1024 blocks

  embed_kernel<<<NPTS, 256, 0, stream>>>(x, W_in1, b_in1, W_in2, b_in2, z);

  for (int l = 0; l < LAYERS; ++l) {
    const size_t wOff = (size_t)l * DMODEL * DMODEL;
    const size_t fOff = (size_t)l * DMODEL * FFD;

    ln_kernel<<<NPTS, 256, 0, stream>>>(z, ln_q_s + l * DMODEL, ln_q_o + l * DMODEL, a_bf);
    wconv_kernel<<<cvt256, 256, 0, stream>>>(Wq + wOff, wT, DMODEL, DMODEL);
    gemm_kernel<0><<<g256, 256, 0, stream>>>(a_bf, wT, bq + l * DMODEL, q_bf, NPTS, DMODEL, DMODEL);

    ln_kernel<<<NPTS, 256, 0, stream>>>(z, ln_k_s + l * DMODEL, ln_k_o + l * DMODEL, a_bf);
    wconv_kernel<<<cvt256, 256, 0, stream>>>(Wk + wOff, wT, DMODEL, DMODEL);
    gemm_kernel<0><<<g256, 256, 0, stream>>>(a_bf, wT, bk + l * DMODEL, k_bf, NPTS, DMODEL, DMODEL);

    ln_kernel<<<NPTS, 256, 0, stream>>>(z, ln_v_s + l * DMODEL, ln_v_o + l * DMODEL, a_bf);
    wconv_kernel<<<cvt256, 256, 0, stream>>>(Wv + wOff, wT, DMODEL, DMODEL);
    gemm_kernel<3><<<g256, 256, 0, stream>>>(a_bf, wT, bv + l * DMODEL, v_bf, NPTS, DMODEL, DMODEL);

    k2_kernel<<<(NPTS * HEADS) / 256, 256, 0, stream>>>(k_bf, k2b);
    attn_kernel<<<(HEADS * NPTS / 16) / 8, 256, 0, stream>>>(q_bf, k_bf, v_bf, k2b, o_bf);

    wconv_kernel<<<cvt256, 256, 0, stream>>>(Wo + wOff, wT, DMODEL, DMODEL);
    gemm_kernel<2><<<g256, 256, 0, stream>>>(o_bf, wT, bo + l * DMODEL, z, NPTS, DMODEL, DMODEL);

    ln_kernel<<<NPTS, 256, 0, stream>>>(z, ln_f_s + l * DMODEL, ln_f_o + l * DMODEL, a_bf);
    wconv_kernel<<<cvt1024, 256, 0, stream>>>(Wf1 + fOff, wT, DMODEL, FFD);   // -> [FFD][DMODEL]
    gemm_kernel<1><<<g1024, 256, 0, stream>>>(a_bf, wT, bf1 + l * FFD, h_bf, NPTS, FFD, DMODEL);
    wconv_kernel<<<cvt1024, 256, 0, stream>>>(Wf2 + fOff, wT, FFD, DMODEL);   // -> [DMODEL][FFD]
    gemm_kernel<2><<<g256, 256, 0, stream>>>(h_bf, wT, bf2 + l * DMODEL, z, NPTS, DMODEL, FFD);
  }

  final_kernel<<<NPTS, 256, 0, stream>>>(z, ln_out_s, ln_out_o, ln_sc_s, ln_sc_o, W_sc, b_sc,
                                         (float*)d_out);
}
